// AttentionPytorch_49409303773427
// MI455X (gfx1250) — matrix operand order repaired and checked by reading, not compiled, not run on hardware
//
#include <hip/hip_runtime.h>

// ---------------------------------------------------------------------------
// CDNA5 (gfx1250, wave32) attention forward:
//   Q,K,V = X @ W{q,k,v}^T   (bf16 storage, fp32 accumulate via v_wmma)
//   A     = (Q K^T) * 1/sqrt(1024)     -> weights region of d_out
//   P     = softmax(A) (in place)
//   ctx   = P @ V                      -> context region of d_out
// GEMM: 128x128 block tile, 8 waves (2M x 4N), 64x32 per wave = 8 WMMA/K-step,
// double-buffered LDS (one barrier per K-step), KS=32.
// ---------------------------------------------------------------------------

typedef __attribute__((ext_vector_type(16))) __bf16 v16bf;
typedef __attribute__((ext_vector_type(8)))  float  v8f;

union Frag16 {            // 16 bf16 = 32 bytes = 8 VGPRs (wave32 WMMA A/B frag)
    v16bf v;
    uint4 q[2];
};

#define TM  128
#define TN  128
#define KS  32
#define AP  40    // a_tile row pitch in bf16 (80B rows, 16B aligned)
#define BP  136   // b_tile row pitch in bf16 (272B rows, 16B aligned)
#define NT  256

// Load 4 contiguous elements as float (fp32 or bf16 source).
template <typename T>
__device__ __forceinline__ void load4f(const T* __restrict__ p, float f[4]) {
    if constexpr (sizeof(T) == 4) {
        const float4 t = *(const float4*)p;
        f[0] = t.x; f[1] = t.y; f[2] = t.z; f[3] = t.w;
    } else {
        union { uint2 u; __bf16 h[4]; } t;
        t.u = *(const uint2*)p;
        f[0] = (float)t.h[0]; f[1] = (float)t.h[1];
        f[2] = (float)t.h[2]; f[3] = (float)t.h[3];
    }
}

// Pack 4 floats -> 4 bf16 and store as one 8-byte LDS write.
__device__ __forceinline__ void store4bf(__bf16* __restrict__ d, const float f[4]) {
    union { uint2 u; __bf16 h[4]; } t;
    t.h[0] = (__bf16)f[0]; t.h[1] = (__bf16)f[1];
    t.h[2] = (__bf16)f[2]; t.h[3] = (__bf16)f[3];
    *(uint2*)d = t.u;
}

// Stage one (A,B) K-step tile pair into LDS buffers (bf16).
template <typename TA, typename TB, bool B_TRANSPOSED>
__device__ __forceinline__ void stage_tiles(
    const TA* __restrict__ A, const TB* __restrict__ B,
    long ab, long bb, int m0, int n0, int K, int N, int kk,
    __bf16* __restrict__ at, __bf16* __restrict__ bt, int tid)
{
    // ---- A tile: at[r][c] = (bf16) A[m0+r][kk+c]   (A row-major [M,K])
#pragma unroll
    for (int it = 0; it < (TM * KS) / (NT * 4); ++it) {
        const int idx = tid * 4 + it * NT * 4;
        const int r = idx >> 5, c = idx & 31;
        const TA* src = A + ab + (long)(m0 + r) * K + kk + c;
        if (it == 0 && kk + 2 * KS < K)
            __builtin_prefetch((const void*)(src + 2 * KS), 0, 3);
        float f[4];
        load4f(src, f);
        store4bf(&at[r * AP + c], f);
    }
    // ---- B tile into bt[k][n]
    if constexpr (B_TRANSPOSED) {
        // B is [N,K] row-major (W / K-matrix): bt[k][n] = B[n0+n][kk+k]
#pragma unroll
        for (int it = 0; it < (TN * KS) / (NT * 4); ++it) {
            const int idx = tid * 4 + it * NT * 4;
            const int n = idx >> 5, c = idx & 31;
            const TB* src = B + bb + (long)(n0 + n) * K + kk + c;
            if (it == 0 && kk + 2 * KS < K)
                __builtin_prefetch((const void*)(src + 2 * KS), 0, 3);
            float f[4];
            load4f(src, f);
#pragma unroll
            for (int e = 0; e < 4; ++e)
                bt[(c + e) * BP + n] = (__bf16)f[e];
        }
    } else {
        // B is [K,N] row-major (V-matrix): bt[k][n] = B[kk+k][n0+n]
#pragma unroll
        for (int it = 0; it < (KS * TN) / (NT * 4); ++it) {
            const int idx = tid * 4 + it * NT * 4;
            const int k = idx >> 7, n = idx & 127;
            const TB* src = B + bb + (long)(kk + k) * N + n0 + n;
            if (it == 0 && kk + 2 * KS < K)
                __builtin_prefetch((const void*)(src + (long)2 * KS * N), 0, 3);
            float f[4];
            load4f(src, f);
            store4bf(&bt[k * BP + n], f);
        }
    }
}

template <typename TA, typename TB, typename TC, bool B_TRANSPOSED>
__global__ __launch_bounds__(NT)
void gemm_wmma_bf16(const TA* __restrict__ A, const TB* __restrict__ B,
                    TC* __restrict__ C, int M, int N, int K,
                    long sA, long sB, long sC, float scale)
{
    __shared__ __align__(16) __bf16 at[2][TM * AP];
    __shared__ __align__(16) __bf16 bt[2][KS * BP];

    const int tid  = threadIdx.x;
    const int lane = tid & 31;
    const int wid  = tid >> 5;
    const int half = lane >> 4;        // lanes 16-31 hold the second K half
    const int l15  = lane & 15;
    const int wm   = (wid >> 2) * 64;  // 2 waves over M, 64 rows each
    const int wn   = (wid & 3) * 32;   // 4 waves over N, 32 cols each

    const long ab = (long)blockIdx.z * sA;
    const long bb = (long)blockIdx.z * sB;
    const long cb = (long)blockIdx.z * sC;
    const int  m0 = blockIdx.y * TM;
    const int  n0 = blockIdx.x * TN;

    v8f acc[4][2];
#pragma unroll
    for (int i = 0; i < 4; ++i)
#pragma unroll
        for (int j = 0; j < 2; ++j)
            acc[i][j] = (v8f){};

    // Prologue: stage first K-step into buffer 0.
    stage_tiles<TA, TB, B_TRANSPOSED>(A, B, ab, bb, m0, n0, K, N, 0,
                                      at[0], bt[0], tid);
    __syncthreads();

    int cur = 0;
    for (int kk = 0; kk < K; kk += KS) {
        // Stage next K-step into the other buffer while computing this one.
        if (kk + KS < K)
            stage_tiles<TA, TB, B_TRANSPOSED>(A, B, ab, bb, m0, n0, K, N,
                                              kk + KS, at[cur ^ 1], bt[cur ^ 1], tid);

        // ---- fragment loads per ISA 7.12.2 layouts
        Frag16 af[4], bf[2];
#pragma unroll
        for (int i = 0; i < 4; ++i) {
            // A 16x32 bf16: lane holds row m=l15; VGPR0-3 = K[half*8..+7],
            // VGPR4-7 = K[16+half*8..+7]  -> two ds_load_b128 per fragment
            const __bf16* ar = &at[cur][(wm + i * 16 + l15) * AP];
            af[i].q[0] = *(const uint4*)(ar + half * 8);
            af[i].q[1] = *(const uint4*)(ar + 16 + half * 8);
        }
#pragma unroll
        for (int j = 0; j < 2; ++j) {
            // B 32x16 bf16: lane holds col n=l15; K[half*8..+7], K[16+half*8..+7]
            const __bf16* br = &bt[cur][wn + j * 16 + l15];
#pragma unroll
            for (int s = 0; s < 8; ++s) {
                bf[j].v[s]     = br[(half * 8 + s) * BP];
                bf[j].v[8 + s] = br[(16 + half * 8 + s) * BP];
            }
        }
#pragma unroll
        for (int i = 0; i < 4; ++i)
#pragma unroll
            for (int j = 0; j < 2; ++j)
                acc[i][j] = __builtin_amdgcn_wmma_f32_16x16x32_bf16(
                    false, af[i].v, false, bf[j].v,
                    (short)0, acc[i][j], false, false);

        __syncthreads();   // next-buffer writes done AND this-buffer reads done
        cur ^= 1;
    }

    // ---- store C: 32-bit D frag: col = n + l15, row = m + half*8 + v
#pragma unroll
    for (int i = 0; i < 4; ++i) {
#pragma unroll
        for (int j = 0; j < 2; ++j) {
            const int rbase = m0 + wm + i * 16 + half * 8;
            const int col   = n0 + wn + j * 16 + l15;
#pragma unroll
            for (int v = 0; v < 8; ++v) {
                const float val = acc[i][j][v] * scale;
                C[cb + (long)(rbase + v) * N + col] = (TC)val;
            }
        }
    }
}

// ---------------------------------------------------------------------------
// Row softmax over 2048 fp32 columns, in place. One 256-thread block per row.
// ---------------------------------------------------------------------------
__global__ __launch_bounds__(256)
void softmax_rows(float* __restrict__ w)
{
    constexpr int NC = 2048;
    float* p = w + (long)blockIdx.x * NC;
    const int t = threadIdx.x, lane = t & 31, wid = t >> 5;

    float v[8];
    float m = -3.402823466e38f;
#pragma unroll
    for (int i = 0; i < 8; ++i) {
        v[i] = p[t + i * 256];
        m = fmaxf(m, v[i]);
    }
#pragma unroll
    for (int off = 16; off > 0; off >>= 1)
        m = fmaxf(m, __shfl_xor(m, off, 32));

    __shared__ float red[8];
    if (lane == 0) red[wid] = m;
    __syncthreads();
    float bm = red[0];
#pragma unroll
    for (int i = 1; i < 8; ++i) bm = fmaxf(bm, red[i]);

    float s = 0.f;
#pragma unroll
    for (int i = 0; i < 8; ++i) {
        v[i] = __expf(v[i] - bm);
        s += v[i];
    }
#pragma unroll
    for (int off = 16; off > 0; off >>= 1)
        s += __shfl_xor(s, off, 32);
    __syncthreads();
    if (lane == 0) red[wid] = s;
    __syncthreads();
    float tot = 0.f;
#pragma unroll
    for (int i = 0; i < 8; ++i) tot += red[i];
    const float inv = 1.0f / tot;
#pragma unroll
    for (int i = 0; i < 8; ++i) p[t + i * 256] = v[i] * inv;
}

// ---------------------------------------------------------------------------
extern "C" void kernel_launch(void* const* d_in, const int* in_sizes, int n_in,
                              void* d_out, int out_size, void* d_ws, size_t ws_size,
                              hipStream_t stream)
{
    (void)in_sizes; (void)n_in; (void)out_size; (void)ws_size;
    constexpr long BATCH = 4, SEQ = 2048, D = 1024;
    constexpr long MTOT = BATCH * SEQ;                 // 8192

    const float* X  = (const float*)d_in[0];
    const float* Wq = (const float*)d_in[1];
    const float* Wk = (const float*)d_in[2];
    const float* Wv = (const float*)d_in[3];

    float* ctx = (float*)d_out;                        // [4,2048,1024]
    float* wts = (float*)d_out + BATCH * SEQ * D;      // [4,2048,2048]

    char* ws = (char*)d_ws;
    __bf16* Qb = reinterpret_cast<__bf16*>(ws);
    __bf16* Kb = reinterpret_cast<__bf16*>(ws + MTOT * D * sizeof(__bf16));
    __bf16* Vb = reinterpret_cast<__bf16*>(ws + 2 * MTOT * D * sizeof(__bf16));

    const dim3 blk(NT);

    // --- QKV projections: [8192,1024] x [1024,1024]^T -> bf16
    const dim3 g1(D / TN, MTOT / TM, 1);
    gemm_wmma_bf16<float, float, __bf16, true><<<g1, blk, 0, stream>>>(
        X, Wq, Qb, (int)MTOT, (int)D, (int)D, 0, 0, 0, 1.0f);
    gemm_wmma_bf16<float, float, __bf16, true><<<g1, blk, 0, stream>>>(
        X, Wk, Kb, (int)MTOT, (int)D, (int)D, 0, 0, 0, 1.0f);
    gemm_wmma_bf16<float, float, __bf16, true><<<g1, blk, 0, stream>>>(
        X, Wv, Vb, (int)MTOT, (int)D, (int)D, 0, 0, 0, 1.0f);

    // --- scores: A = Q K^T * (1/sqrt(1024)), batched over 4
    const dim3 g2(SEQ / TN, SEQ / TM, BATCH);
    gemm_wmma_bf16<__bf16, __bf16, float, true><<<g2, blk, 0, stream>>>(
        Qb, Kb, wts, (int)SEQ, (int)SEQ, (int)D,
        SEQ * D, SEQ * D, SEQ * SEQ, 0.03125f);

    // --- softmax over last dim, in place
    softmax_rows<<<dim3((unsigned)(BATCH * SEQ)), blk, 0, stream>>>(wts);

    // --- context = P @ V  (P fp32 -> bf16 on the fly; V is [K=2048, N=1024])
    const dim3 g3(D / TN, SEQ / TM, BATCH);
    gemm_wmma_bf16<float, __bf16, float, false><<<g3, blk, 0, stream>>>(
        wts, Vb, ctx, (int)SEQ, (int)D, (int)SEQ,
        SEQ * SEQ, SEQ * D, SEQ * D, 1.0f);
}
